// ReLUKAN_55860344652229
// MI455X (gfx1250) — matrix operand order repaired and checked by reading, not compiled, not run on hardware
//
#include <hip/hip_runtime.h>
#include <math.h>

// ---------------------------------------------------------------------------
// ReLU-KAN forward for MI455X (gfx1250, wave32, WMMA).
//   Hot path: 7 GEMMs  M=2048, N=256, K=16384  with A = relu(z - grid)
//   generated in registers and computed via v_wmma_f32_16x16x32_bf16.
//   fs (117MB f32, L2-resident) is pre-packed once per launch into bf16 tiles
//   (pre-scaled by the depth coefficient) matching the B-operand register
//   image, so the inner loop loads B with one coalesced 32B/lane vector load.
// ---------------------------------------------------------------------------

typedef __attribute__((ext_vector_type(16))) __bf16 v16bf;
typedef __attribute__((ext_vector_type(2)))  __bf16 v2bf;
typedef __attribute__((ext_vector_type(8)))  float  v8f;
typedef __attribute__((ext_vector_type(16))) float  v16f;
typedef __attribute__((ext_vector_type(2)))  float  v2f;

#define BATCH  2048
#define IN_DIM 128
#define GRID_N 128
#define HID    256
#define DEPTH  7
#define KTOT   (IN_DIM * GRID_N)      // 16384 per depth
#define KCHUNKS 512                   // 16384 / 32
#define NTILES  16                    // 256 / 16

// f32 pair -> packed bf16 via vector fptrunc (backend selects hw converter)
__device__ __forceinline__ unsigned pack_bf16(float a, float b) {
  union { v2bf v; unsigned u; } cv;
  v2f t = {a, b};
  cv.v = __builtin_convertvector(t, v2bf);
  return cv.u;
}

// ---------------------------------------------------------------------------
// 0) depth coefficients: cd[d] = sum_{e>=d} softmax(scores)_e
//    (residual accumulation + depth attention folded into one scale per depth)
// ---------------------------------------------------------------------------
__global__ void coef_kernel(const float* __restrict__ scores, float* __restrict__ cd) {
  if (threadIdx.x == 0 && blockIdx.x == 0) {
    float s[DEPTH]; float mx = -1e30f;
    for (int e = 0; e < DEPTH; ++e) { s[e] = scores[e]; mx = fmaxf(mx, s[e]); }
    float den = 0.f;
    for (int e = 0; e < DEPTH; ++e) { s[e] = __expf(s[e] - mx); den += s[e]; }
    float suf = 0.f;
    for (int d = DEPTH - 1; d >= 0; --d) { suf += s[d]; cd[d] = suf / den; }
  }
}

// ---------------------------------------------------------------------------
// 1) per-feature min/max over batch, write transposed z:  zT[i][b]
// ---------------------------------------------------------------------------
__global__ void minmax_z(const float* __restrict__ x, float* __restrict__ zT) {
  __shared__ float smin[256], smax[256];
  const int i = blockIdx.x;           // feature
  const int tid = threadIdx.x;
  float mn = 1e30f, mx = -1e30f;
  for (int b = tid; b < BATCH; b += 256) {
    float v = x[b * IN_DIM + i];
    mn = fminf(mn, v); mx = fmaxf(mx, v);
  }
  smin[tid] = mn; smax[tid] = mx;
  __syncthreads();
  for (int s = 128; s > 0; s >>= 1) {
    if (tid < s) {
      smin[tid] = fminf(smin[tid], smin[tid + s]);
      smax[tid] = fmaxf(smax[tid], smax[tid + s]);
    }
    __syncthreads();
  }
  const float xmin = smin[0];
  const float inv  = 1.0f / (smax[0] - smin[0] + 1e-6f);
  for (int b = tid; b < BATCH; b += 256)
    zT[(long)i * BATCH + b] = (x[b * IN_DIM + i] - xmin) * inv;
}

// ---------------------------------------------------------------------------
// 2) pack fs f32 -> bf16 (pre-scaled by cd[d]) into per-tile B register image.
//    Tile (d, kc, nt) = 32(K) x 16(N), 1KB: element index = lane*16 + v*2 + p,
//    lane's column n = nt*16 + (lane&15), row kk = (lane>>4)*8 + (v>>2)*16 + (v&3)*2 + p
//    (K 0-7/8-15 split across lane halves, +16 for v>=4; same order as A).
// ---------------------------------------------------------------------------
__global__ void pack_fs(const float* __restrict__ fs, const float* __restrict__ cdbuf,
                        unsigned* __restrict__ dst) {
  const long ppos = (long)blockIdx.x * 256 + threadIdx.x;  // 14,680,064 total
  const int  v    = (int)(ppos & 7);
  const int  lane = (int)((ppos >> 3) & 31);
  const long tile = ppos >> 8;
  const int  nt = (int)(tile & 15);
  const int  kc = (int)((tile >> 4) & 511);
  const int  d  = (int)(tile >> 13);
  const float cd = cdbuf[d];
  const int  kk = ((lane >> 4) << 3) + ((v >> 2) << 4) + (v & 3) * 2;
  const long k  = (long)kc * 32 + kk;
  const int  n  = nt * 16 + (lane & 15);
  const long src = ((long)d * KTOT + k) * HID + n;
  dst[ppos] = pack_bf16(fs[src] * cd, fs[src + HID] * cd);   // k, k+1 pair
}

// ---------------------------------------------------------------------------
// 3) main KAN GEMM.  grid = (16 Mblk, 2 Nblk, 7 depth), block = 256 (8 waves).
//    Wave tile = 32(M) x 64(N): 2x4 accumulators, 8 WMMA per 32-wide K chunk.
//    A layout per ISA 7.12.2 (K pairs split 0-7/8-15 across lane halves, +16
//    for VGPRs 4..7).  A-vector element e = 2v+p matches g[] index order.
// ---------------------------------------------------------------------------
__global__ __launch_bounds__(256) void kan_gemm(
    const float* __restrict__ zT,
    const float* __restrict__ grids,
    const unsigned short* __restrict__ fsb,
    float* __restrict__ partial) {
  const int d = blockIdx.z;
  const int lane = threadIdx.x & 31;
  const int wv   = threadIdx.x >> 5;
  const int wm = wv & 3, wn = wv >> 2;
  const int m0  = blockIdx.x * 128 + wm * 32;       // wave's first row
  const int nt0 = blockIdx.y * 8 + wn * 4;          // wave's first 16-col tile
  const int hl = lane >> 4;
  const int lm = lane & 15;

  const v8f vzero = {0.f, 0.f, 0.f, 0.f, 0.f, 0.f, 0.f, 0.f};
  v8f acc[2][4];
#pragma unroll
  for (int t = 0; t < 2; ++t)
#pragma unroll
    for (int j = 0; j < 4; ++j) acc[t][j] = vzero;

  const unsigned short* tb0 =
      fsb + (((long)d * KCHUNKS * NTILES + nt0) << 9) + lane * 16;

  for (int i = 0; i < IN_DIM; ++i) {                 // feature = K/128
    const float z0 = zT[(long)i * BATCH + m0 + lm];  // coalesced (transposed z)
    const float z1 = zT[(long)i * BATCH + m0 + 16 + lm];
    const float* gip = grids + ((long)d * IN_DIM + i) * GRID_N;
#pragma unroll
    for (int gc = 0; gc < 4; ++gc) {                 // 4 chunks of K=32 per i
      const float* gp = gip + gc * 32;
      // grid values for this lane's K slots (uniform across halves -> bcast)
      const float4 ga = *(const float4*)(gp + hl * 8);
      const float4 gb = *(const float4*)(gp + hl * 8 + 4);
      const float4 gc4 = *(const float4*)(gp + 16 + hl * 8);
      const float4 gd = *(const float4*)(gp + 16 + hl * 8 + 4);
      const float g[16] = {ga.x, ga.y, ga.z, ga.w, gb.x, gb.y, gb.z, gb.w,
                           gc4.x, gc4.y, gc4.z, gc4.w, gd.x, gd.y, gd.z, gd.w};
      v16f a0f, a1f;
#pragma unroll
      for (int e = 0; e < 16; ++e) {
        a0f[e] = fmaxf(z0 - g[e], 0.f);
        a1f[e] = fmaxf(z1 - g[e], 0.f);
      }
      const v16bf A0 = __builtin_convertvector(a0f, v16bf);
      const v16bf A1 = __builtin_convertvector(a1f, v16bf);
      const unsigned short* tb = tb0 + ((long)(i * 4 + gc) * NTILES << 9);
#pragma unroll
      for (int j = 0; j < 4; ++j) {
        const v16bf B = *(const v16bf*)(tb + (j << 9));   // 32B/lane, coalesced
        acc[0][j] = __builtin_amdgcn_wmma_f32_16x16x32_bf16(
            false, A0, false, B, (short)0, acc[0][j], false, false);
        acc[1][j] = __builtin_amdgcn_wmma_f32_16x16x32_bf16(
            false, A1, false, B, (short)0, acc[1][j], false, false);
      }
    }
  }
  // C/D layout: VGPR r -> M = r + 8*(lane>=16), N = lane&15
  float* op = partial + (long)d * BATCH * HID;
#pragma unroll
  for (int t = 0; t < 2; ++t) {
    const int mb = m0 + t * 16 + hl * 8;
#pragma unroll
    for (int j = 0; j < 4; ++j) {
      const int n = (nt0 + j) * 16 + lm;
#pragma unroll
      for (int r = 0; r < 8; ++r)
        op[(long)(mb + r) * HID + n] = acc[t][j][r];
    }
  }
}

// ---------------------------------------------------------------------------
// 4) deterministic depth reduction: h = sum_d partial[d]
// ---------------------------------------------------------------------------
__global__ void depth_sum(const float* __restrict__ partial, float* __restrict__ h) {
  const long idx = (long)blockIdx.x * 256 + threadIdx.x;   // 524288 total
  float s = 0.f;
#pragma unroll
  for (int d = 0; d < DEPTH; ++d) s += partial[(long)d * (BATCH * HID) + idx];
  h[idx] = s;
}

// ---------------------------------------------------------------------------
// 5) head GEMM: y[b][o] = sum_k h[b][k] * W[k][o] + bias[o]
// ---------------------------------------------------------------------------
__global__ void head_gemm(const float* __restrict__ hin, const float* __restrict__ W,
                          const float* __restrict__ bias, float* __restrict__ hout) {
  __shared__ float row[HID];
  const int b = blockIdx.x, o = threadIdx.x;
  row[o] = hin[(long)b * HID + o];
  __syncthreads();
  float a = bias[o];
#pragma unroll 8
  for (int k = 0; k < HID; ++k) a = fmaf(row[k], W[(long)k * HID + o], a);
  hout[(long)b * HID + o] = a;
}

// ---------------------------------------------------------------------------
// 6) batch-stat BatchNorm (biased var) + exact erf GELU, in place, per column
// ---------------------------------------------------------------------------
__global__ void bn_gelu(float* __restrict__ y, const float* __restrict__ gamma,
                        const float* __restrict__ beta) {
  __shared__ float ssum[256], ssq[256];
  const int o = blockIdx.x, tid = threadIdx.x;
  float s = 0.f, q = 0.f;
  for (int b = tid; b < BATCH; b += 256) {
    float v = y[(long)b * HID + o];
    s += v; q += v * v;
  }
  ssum[tid] = s; ssq[tid] = q;
  __syncthreads();
  for (int st = 128; st > 0; st >>= 1) {
    if (tid < st) { ssum[tid] += ssum[tid + st]; ssq[tid] += ssq[tid + st]; }
    __syncthreads();
  }
  const float mean = ssum[0] * (1.0f / BATCH);
  const float var  = ssq[0] * (1.0f / BATCH) - mean * mean;   // biased, torch BN
  const float sc = rsqrtf(var + 1e-5f) * gamma[o];
  const float sh = beta[o];
  for (int b = tid; b < BATCH; b += 256) {
    float v = (y[(long)b * HID + o] - mean) * sc + sh;
    v = 0.5f * v * (1.0f + erff(v * 0.70710678118654752f));   // exact GELU
    y[(long)b * HID + o] = v;
  }
}

// ---------------------------------------------------------------------------
// 7) final GEMV: out[b] = dot(h[b], W_out) + b_out
// ---------------------------------------------------------------------------
__global__ void final_out(const float* __restrict__ h, const float* __restrict__ Wout,
                          const float* __restrict__ bout, float* __restrict__ out) {
  __shared__ float s[256];
  const int b = blockIdx.x, tid = threadIdx.x;
  s[tid] = h[(long)b * HID + tid] * Wout[tid];
  __syncthreads();
  for (int st = 128; st > 0; st >>= 1) {
    if (tid < st) s[tid] += s[tid + st];
    __syncthreads();
  }
  if (tid == 0) out[b] = s[0] + bout[0];
}

// ---------------------------------------------------------------------------
extern "C" void kernel_launch(void* const* d_in, const int* in_sizes, int n_in,
                              void* d_out, int out_size, void* d_ws, size_t ws_size,
                              hipStream_t stream) {
  (void)in_sizes; (void)n_in; (void)out_size; (void)ws_size;
  const float* x      = (const float*)d_in[0];
  const float* grids  = (const float*)d_in[1];
  const float* fs     = (const float*)d_in[2];
  const float* scores = (const float*)d_in[3];
  const float* mlp_W  = (const float*)d_in[4];
  const float* mlp_b  = (const float*)d_in[5];
  const float* gamma  = (const float*)d_in[6];
  const float* beta   = (const float*)d_in[7];
  const float* W_out  = (const float*)d_in[8];
  const float* b_out  = (const float*)d_in[9];
  float* out = (float*)d_out;

  // workspace layout (256B-aligned offsets); total ~78.7 MB
  char* ws = (char*)d_ws;
  float*          zT      = (float*)(ws);                              //  1.0 MB
  unsigned short* fsb     = (unsigned short*)(ws + 1048576u);          // 56.0 MB
  float*          partial = (float*)(ws + 59768832u);                  // 14.0 MB
  float*          h0      = (float*)(ws + 74448896u);                  //  2.0 MB
  float*          h1      = (float*)(ws + 76546048u);                  //  2.0 MB
  float*          cdbuf   = (float*)(ws + 78643200u);                  //  28 B

  coef_kernel<<<1, 32, 0, stream>>>(scores, cdbuf);
  minmax_z<<<128, 256, 0, stream>>>(x, zT);
  pack_fs<<<57344, 256, 0, stream>>>(fs, cdbuf, (unsigned*)fsb);
  kan_gemm<<<dim3(16, 2, DEPTH), 256, 0, stream>>>(zT, grids, fsb, partial);
  depth_sum<<<2048, 256, 0, stream>>>(partial, h0);

  float* cur = h0; float* nxt = h1;
  for (int j = 0; j < 3; ++j) {
    head_gemm<<<2048, 256, 0, stream>>>(cur, mlp_W + (long)j * HID * HID,
                                        mlp_b + j * HID, nxt);
    bn_gelu<<<256, 256, 0, stream>>>(nxt, gamma + j * HID, beta + j * HID);
    float* t = cur; cur = nxt; nxt = t;
  }
  final_out<<<2048, 256, 0, stream>>>(cur, W_out, b_out, out);
}
